// StructureLearner_1778116461065
// MI455X (gfx1250) — matrix operand order repaired and checked by reading, not compiled, not run on hardware
//
#include <hip/hip_runtime.h>
#include <math.h>

#define B_BATCH 4096
#define KNBR    64
#define FDIM    128
#define TEDIM   32
#define DDIM    160
#define NPROJ   320            // stacked [Wk ; Wv] output rows
#define KV_STRIDE 168          // u16 stride of kv_lds rows (336 B, 16B aligned)
#define WSTR      168          // u16 stride of staged weight rows (336 B)
#define OUT_STRIDE 328         // f32 stride of kvout rows
#define WBUF_BYTES (NPROJ * WSTR * 2)   // 107520 B, also holds kvout (83968 B)

typedef unsigned int u32;
typedef unsigned short u16;
typedef unsigned long long u64;
typedef __attribute__((ext_vector_type(4)))  u32    u32x4;
typedef __attribute__((ext_vector_type(8)))  u32    u32x8;
typedef __attribute__((ext_vector_type(16))) __bf16 bf16x16;
typedef __attribute__((ext_vector_type(8)))  float  f32x8;

__device__ __forceinline__ u16 f32_to_bf16(float f) {
  u32 u = __float_as_uint(f);
  u32 r = u + 0x7FFFu + ((u >> 16) & 1u);   // round-to-nearest-even
  return (u16)(r >> 16);
}

// Pack two f32 -> two bf16 in one u32; prefer the HW packed convert.
#if __has_builtin(__builtin_amdgcn_cvt_pk_bf16_f32)
__device__ __forceinline__ u32 pack2_bf16(float a, float b) {
  auto r = __builtin_amdgcn_cvt_pk_bf16_f32(a, b);
  return __builtin_bit_cast(u32, r);
}
#else
__device__ __forceinline__ u32 pack2_bf16(float a, float b) {
  return (u32)f32_to_bf16(a) | ((u32)f32_to_bf16(b) << 16);
}
#endif

// Low 32 bits of a flat LDS pointer are the LDS byte offset (ISA 10.2:
// LDS aperture keeps the offset in addr[31:0]).
__device__ __forceinline__ u32 lds_byte_offset(const void* p) {
  return (u32)(size_t)p;
}

// Convert stacked [Wk ; Wv] = in_proj_w rows 160..479 (320 x 160) to bf16.
__global__ void prep_weights_bf16(const float* __restrict__ in_proj_w,
                                  u16* __restrict__ wkv) {
  int i = blockIdx.x * blockDim.x + threadIdx.x;
  if (i < NPROJ * DDIM) wkv[i] = f32_to_bf16(in_proj_w[DDIM * DDIM + i]);
}

__global__ void __launch_bounds__(256) fused_tga(
    const float* __restrict__ x,
    const int*   __restrict__ tgt_ids,
    const int*   __restrict__ tgt_times,
    const int*   __restrict__ nbr_ids,
    const int*   __restrict__ edge_time,
    const float* __restrict__ edge_weight,
    const float* __restrict__ gumbel_u,
    const float* __restrict__ te_w,
    const float* __restrict__ te_b,
    const float* __restrict__ in_proj_w,
    const float* __restrict__ in_proj_b,
    const float* __restrict__ out_w,
    const float* __restrict__ out_b,
    const float* __restrict__ mlp_w,
    const float* __restrict__ mlp_b,
    const u16*   __restrict__ wkv_bf16,
    float* __restrict__ out_attn,
    float* __restrict__ out_new_w,
    float* __restrict__ out_mask)
{
  const int b = blockIdx.x;
  const int t = threadIdx.x;

  __shared__ __align__(16) u16 kv_lds[KNBR * KV_STRIDE];   // kv_in, bf16
  // Union region: bf16 weight stage during WMMA phase, then kvout (f32 [k|v]).
  __shared__ __align__(16) unsigned char ubuf[WBUF_BYTES];
  __shared__ float q_in[DDIM];
  __shared__ float qv[DDIM];
  __shared__ float sc[KNBR];
  __shared__ float attnw[KNBR];
  __shared__ float ctxs[DDIM];
  __shared__ float aout[DDIM];
  __shared__ float zbuf[KNBR];
  __shared__ float smlp_s;

  u16*   wbuf  = (u16*)ubuf;
  float* kvout = (float*)ubuf;

  // ---------- stage 1a: async-stage bf16 weights global -> LDS ----------
  // 320 rows x 320 B, row stride in LDS = 336 B. 6400 x 16B chunks, 25/thread.
  {
    const u64 gbase = (u64)(size_t)wkv_bf16;
    const u32 lbase = lds_byte_offset(wbuf);
    #pragma unroll
    for (int c = 0; c < 25; ++c) {
      const int chunk = t + c * 256;          // 0..6399
      const int row   = chunk / 20;
      const int piece = chunk % 20;
      const u32 gofs = (u32)(row * 320 + piece * 16);
      const u32 lofs = lbase + (u32)(row * 336 + piece * 16);
      asm volatile("global_load_async_to_lds_b128 %0, %1, %2"
                   :: "v"(lofs), "v"(gofs), "s"(gbase)
                   : "memory");
    }
  }

  // ---------- stage 1b: gather kv_in (bf16 -> LDS) and q_in (f32) ----------
  {
    const int j = t >> 2;                 // neighbor row 0..63
    const int p = t & 3;                  // quarter of the row
    const int nid = nbr_ids[b * KNBR + j];
    const float4* xr = (const float4*)(x + (size_t)nid * FDIM + p * 32);
    #pragma unroll
    for (int c = 0; c < 8; ++c) {
      float4 v4 = xr[c];
      u32* dst = (u32*)&kv_lds[j * KV_STRIDE + p * 32 + c * 4];
      dst[0] = pack2_bf16(v4.x, v4.y);
      dst[1] = pack2_bf16(v4.z, v4.w);
    }
    const float et = (float)edge_time[b * KNBR + j];
    #pragma unroll
    for (int i = 0; i < 8; i += 2) {
      int c = p * 8 + i;
      float c0 = cosf(et * te_w[c] + te_b[c]);
      float c1 = cosf(et * te_w[c + 1] + te_b[c + 1]);
      *(u32*)&kv_lds[j * KV_STRIDE + FDIM + c] = pack2_bf16(c0, c1);
    }
  }
  if (t < DDIM) {
    if (t < FDIM) {
      q_in[t] = x[(size_t)tgt_ids[b] * FDIM + t];
    } else {
      const float tt = (float)tgt_times[b];
      q_in[t] = cosf(tt * te_w[t - FDIM] + te_b[t - FDIM]);
    }
  }
  asm volatile("s_wait_asynccnt 0x0" ::: "memory");   // weight stage landed
  __syncthreads();

  // ---------- stage 2: [k|v] = kv_in * [Wk;Wv]^T via bf16 WMMA ----------
  const int lane  = t & 31;
  const int w     = t >> 5;       // wave 0..7
  const int h     = lane >> 4;    // half-wave
  const int ln    = lane & 15;
  const int mtile = w & 3;        // M tile 0..3 (rows of 16)
  const int nhalf = w >> 2;       // N half: 10 tiles each

  f32x8 acc[10] = {};
  {
    const int arow = mtile * 16 + ln;
    for (int ks = 0; ks < 5; ++ks) {
      const int k0 = ks * 32;
      // A fragment: row arow, K chunks [k0+8h, +8) and [k0+16+8h, +8)
      u32x4 alo = *(const u32x4*)&kv_lds[arow * KV_STRIDE + k0 + 8 * h];
      u32x4 ahi = *(const u32x4*)&kv_lds[arow * KV_STRIDE + k0 + 16 + 8 * h];
      u32x8 au;
      au[0] = alo[0]; au[1] = alo[1]; au[2] = alo[2]; au[3] = alo[3];
      au[4] = ahi[0]; au[5] = ahi[1]; au[6] = ahi[2]; au[7] = ahi[3];
      bf16x16 afrag = __builtin_bit_cast(bf16x16, au);

      // Preload all 10 B fragments into distinct registers, then issue the
      // 10 WMMAs back-to-back (no D->A/B overlap => no hazard NOPs).
      bf16x16 bfr[10];
      #pragma unroll
      for (int nt = 0; nt < 10; ++nt) {
        const int ncol = (nhalf * 10 + nt) * 16 + ln;
        const u32x4* bp = (const u32x4*)&wbuf[ncol * WSTR + k0 + 8 * h];
        u32x4 b0 = bp[0], b1 = bp[2];
        u32x8 bu;
        bu[0] = b0[0]; bu[1] = b0[1]; bu[2] = b0[2]; bu[3] = b0[3];
        bu[4] = b1[0]; bu[5] = b1[1]; bu[6] = b1[2]; bu[7] = b1[3];
        bfr[nt] = __builtin_bit_cast(bf16x16, bu);
      }
      #pragma unroll
      for (int nt = 0; nt < 10; ++nt) {
        acc[nt] = __builtin_amdgcn_wmma_f32_16x16x32_bf16(
            false, afrag, false, bfr[nt], (short)0, acc[nt], false, false);
      }
    }
  }
  // q = q_in * Wq^T + bq  (full f32 for score accuracy)
  if (t < DDIM) {
    float a = in_proj_b[t];
    const float* wr = in_proj_w + (size_t)t * DDIM;
    for (int f = 0; f < DDIM; ++f) a += q_in[f] * wr[f];
    qv[t] = a;
  }
  __syncthreads();   // all waves done READING wbuf; accumulators live in regs

  // ---------- stage 3: store C tiles (+bias) into kvout (overlays wbuf) ----
  {
    #pragma unroll
    for (int nt = 0; nt < 10; ++nt) {
      const int ncol = (nhalf * 10 + nt) * 16 + ln;
      const float bias = in_proj_b[DDIM + ncol];
      #pragma unroll
      for (int r = 0; r < 8; ++r) {
        kvout[(mtile * 16 + 8 * h + r) * OUT_STRIDE + ncol] = acc[nt][r] + bias;
      }
    }
  }
  __syncthreads();

  // ---------- stage 4: scores = q . k / sqrt(D) ----------
  {
    const int j = t >> 2;
    const int p = t & 3;
    float part = 0.f;
    const float* kr = &kvout[j * OUT_STRIDE];
    for (int d = p * 40; d < p * 40 + 40; ++d) part += qv[d] * kr[d];
    part += __shfl_xor(part, 1, 32);
    part += __shfl_xor(part, 2, 32);
    if (p == 0) sc[j] = part * 0.07905694150420949f;   // 1/sqrt(160)
  }
  __syncthreads();

  // ---------- stage 5: softmax over K ----------
  if (t < KNBR) {
    float mx = -INFINITY;
    for (int j = 0; j < KNBR; ++j) mx = fmaxf(mx, sc[j]);
    float sum = 0.f;
    for (int j = 0; j < KNBR; ++j) sum += expf(sc[j] - mx);
    attnw[t] = expf(sc[t] - mx) / sum;
  }
  __syncthreads();

  // ---------- stage 6: ctx = attn_w . v ----------
  if (t < DDIM) {
    float a = 0.f;
    for (int j = 0; j < KNBR; ++j) a += attnw[j] * kvout[j * OUT_STRIDE + DDIM + t];
    ctxs[t] = a;
  }
  __syncthreads();

  // ---------- stage 7: attn_output = ctx * out_w^T + out_b ----------
  if (t < DDIM) {
    float a = out_b[t];
    const float* wr = out_w + (size_t)t * DDIM;
    for (int e = 0; e < DDIM; ++e) a += ctxs[e] * wr[e];
    aout[t] = a;
    out_attn[(size_t)b * DDIM + t] = a;
  }
  __syncthreads();

  // ---------- stage 8: smlp = attn_output . mlp_w[:D]; gumbel logits ------
  if (t < 32) {
    float p = 0.f;
    #pragma unroll
    for (int i = 0; i < 5; ++i) p += aout[t * 5 + i] * mlp_w[t * 5 + i];
    p += __shfl_xor(p, 16, 32);
    p += __shfl_xor(p, 8, 32);
    p += __shfl_xor(p, 4, 32);
    p += __shfl_xor(p, 2, 32);
    p += __shfl_xor(p, 1, 32);
    if (t == 0) smlp_s = p;
  }
  if (t < KNBR) {
    const float u = gumbel_u[b * KNBR + t];
    const float g = -logf(-logf(u + 1e-10f) + 1e-10f);
    zbuf[t] = attnw[t] + g;                 // TAU = 1
  }
  __syncthreads();

  // ---------- stage 9: gumbel softmax mask + edge MLP ----------
  if (t < KNBR) {
    float mz = -INFINITY;
    for (int j = 0; j < KNBR; ++j) mz = fmaxf(mz, zbuf[j]);
    float sz = 0.f;
    for (int j = 0; j < KNBR; ++j) sz += expf(zbuf[j] - mz);
    const float y = expf(zbuf[t] - mz) / sz;
    out_mask[(size_t)b * KNBR + t] = (y > 0.2f) ? 1.0f : 0.0f;

    const float ew = edge_weight[b * KNBR + t];
    float val = smlp_s + ew * mlp_w[DDIM] + mlp_b[0];
    out_new_w[(size_t)b * KNBR + t] = (val >= 0.f) ? val : 0.01f * val;
  }
}

extern "C" void kernel_launch(void* const* d_in, const int* in_sizes, int n_in,
                              void* d_out, int out_size, void* d_ws, size_t ws_size,
                              hipStream_t stream) {
  const float* x        = (const float*)d_in[0];
  const int*   tgt_ids  = (const int*)d_in[1];
  const int*   tgt_tms  = (const int*)d_in[2];
  const int*   nbr_ids  = (const int*)d_in[3];
  const int*   etime    = (const int*)d_in[4];
  const float* eweight  = (const float*)d_in[5];
  const float* gumbel   = (const float*)d_in[6];
  const float* te_w     = (const float*)d_in[7];
  const float* te_b     = (const float*)d_in[8];
  const float* ipw      = (const float*)d_in[9];
  const float* ipb      = (const float*)d_in[10];
  const float* ow       = (const float*)d_in[11];
  const float* ob       = (const float*)d_in[12];
  const float* mw       = (const float*)d_in[13];
  const float* mb       = (const float*)d_in[14];

  u16*   wkv      = (u16*)d_ws;
  float* out      = (float*)d_out;
  float* out_attn = out;
  float* out_new  = out + (size_t)B_BATCH * DDIM;
  float* out_mask = out_new + (size_t)B_BATCH * KNBR;

  prep_weights_bf16<<<(NPROJ * DDIM + 255) / 256, 256, 0, stream>>>(ipw, wkv);
  fused_tga<<<B_BATCH, 256, 0, stream>>>(x, tgt_ids, tgt_tms, nbr_ids, etime,
                                         eweight, gumbel, te_w, te_b, ipw, ipb,
                                         ow, ob, mw, mb, wkv,
                                         out_attn, out_new, out_mask);
  (void)in_sizes; (void)n_in; (void)out_size; (void)ws_size;
}
